// DMPNNNet_42399917146353
// MI455X (gfx1250) — matrix operand order repaired and checked by reading, not compiled, not run on hardware
//
#include <hip/hip_runtime.h>
#include <hip/hip_bf16.h>

typedef __attribute__((ext_vector_type(16))) _Float16 v16h;
typedef __attribute__((ext_vector_type(8)))  float    v8f;

#define KPAD_MAX 160          // max K (142 -> padded to 5*32)
#define WSLOT_HALF (KPAD_MAX*64)

// ---------------- small utility kernels ----------------

__global__ void k_zero_f32(float* p, long long n) {
  long long t = (long long)blockIdx.x * blockDim.x + threadIdx.x;
  if (t < n) p[t] = 0.f;
}

static __device__ __forceinline__ unsigned long long mix64(unsigned long long x) {
  x += 0x9e3779b97f4a7c15ull;
  x = (x ^ (x >> 30)) * 0xbf58476d1ce4e5b9ull;
  x = (x ^ (x >> 27)) * 0x94d049bb133111ebull;
  return x ^ (x >> 31);
}

__global__ void k_hash_clear(unsigned long long* keys, int hs) {
  int t = blockIdx.x * blockDim.x + threadIdx.x;
  if (t < hs) keys[t] = ~0ull;
}

__global__ void k_hash_insert(const int* row, const int* col, long long E, long long N,
                              unsigned long long* keys, int* vals, unsigned int mask) {
  long long e = (long long)blockIdx.x * blockDim.x + threadIdx.x;
  if (e >= E) return;
  unsigned long long key = (unsigned long long)(unsigned)row[e] * (unsigned long long)N + (unsigned)col[e];
  unsigned int h = (unsigned int)mix64(key) & mask;
  for (;;) {
    unsigned long long prev = atomicCAS(&keys[h], ~0ull, key);
    if (prev == ~0ull || prev == key) { vals[h] = (int)e; break; }
    h = (h + 1) & mask;
  }
}

__global__ void k_hash_lookup(const int* row, const int* col, long long E, long long N,
                              const unsigned long long* keys, const int* vals, unsigned int mask,
                              int* rev, int* hasrev) {
  long long e = (long long)blockIdx.x * blockDim.x + threadIdx.x;
  if (e >= E) return;
  unsigned long long rkey = (unsigned long long)(unsigned)col[e] * (unsigned long long)N + (unsigned)row[e];
  unsigned int h = (unsigned int)mix64(rkey) & mask;
  for (;;) {
    unsigned long long k = keys[h];
    if (k == rkey)  { rev[e] = vals[h]; hasrev[e] = 1; return; }
    if (k == ~0ull) { rev[e] = 0;       hasrev[e] = 0; return; }
    h = (h + 1) & mask;
  }
}

// eh0 = edge_attr @ We + be   ([E,6] x [6,64])
__global__ void k_edge_init(const float* __restrict__ ea, const float* __restrict__ We,
                            const float* __restrict__ be, float* __restrict__ eh, long long E) {
  long long t = (long long)blockIdx.x * blockDim.x + threadIdx.x;
  if (t >= E * 64) return;
  long long e = t >> 6; int hcol = (int)(t & 63);
  float s = be[hcol];
  const float* a = ea + e * 6;
  #pragma unroll
  for (int k = 0; k < 6; ++k) s += a[k] * We[k * 64 + hcol];
  eh[t] = s;
}

// dst[idx[r]] += src[r]  for [R,64] rows
__global__ void k_segsum64(const float* __restrict__ src, const int* __restrict__ idx,
                           float* __restrict__ dst, long long R) {
  long long t = (long long)blockIdx.x * blockDim.x + threadIdx.x;
  if (t >= R * 64) return;
  long long r = t >> 6; int hcol = (int)(t & 63);
  atomicAdd(&dst[(long long)idx[r] * 64 + hcol], src[t]);
}

// Convert f32 weight [Kin,64] into WMMA-B-fragment-swizzled f16, zero-padded to KC*32.
// dst layout: (((kc*4+nt)*32+lane)*16 + i) with k = kc*32 + (lane>=16?16:0) + i, n = nt*16 + (lane&15)
__global__ void k_prep_w(const float* __restrict__ W, int Kin, int KC, _Float16* __restrict__ dst) {
  int t = blockIdx.x * blockDim.x + threadIdx.x;
  int total = KC * 2048;
  if (t >= total) return;
  int i    = t & 15;
  int lane = (t >> 4) & 31;
  int nt   = (t >> 9) & 3;
  int kc   = t >> 11;
  int k = kc * 32 + ((lane >= 16) ? 16 : 0) + i;
  int n = nt * 16 + (lane & 15);
  float v = (k < Kin) ? W[k * 64 + n] : 0.f;
  dst[t] = (_Float16)v;
}

// ---------------- WMMA tile engine ----------------

union AF { v16h v; _Float16 h[16]; };
union BF { v16h v; _Float16 h[16]; uint4 q[2]; };

// One wave computes a 16-row x 64-col GEMM tile from LDS-staged A (f16, stride KPAD_MAX)
// and pre-swizzled LDS weights, then stores relu(acc+bias) to out (row stride 64).
static __device__ __forceinline__ void wmma_rows16(const _Float16* __restrict__ sAw,
                                                   const _Float16* __restrict__ sW, int KC,
                                                   const float* __restrict__ bias,
                                                   float* __restrict__ out,
                                                   long long rowBase, long long nRows, int lane) {
  v8f acc[4] = {};
  const int m  = lane & 15;
  const int hi = lane >> 4;
  for (int kc = 0; kc < KC; ++kc) {
    AF a;
    const _Float16* ar = sAw + m * KPAD_MAX + kc * 32;
    #pragma unroll
    for (int v = 0; v < 8; ++v) {
      // A k order used here (contiguous): lanes<16: K=0..15 ; lanes>=16: K=16..31 (matches B swizzle)
      int k = (hi << 4) + (v << 1);
      a.h[2 * v]     = ar[k];
      a.h[2 * v + 1] = ar[k + 1];
    }
    #pragma unroll
    for (int nt = 0; nt < 4; ++nt) {
      BF b;
      const uint4* bp = (const uint4*)(sW + ((size_t)((kc * 4 + nt) * 32 + lane)) * 16);
      b.q[0] = bp[0];
      b.q[1] = bp[1];
      acc[nt] = __builtin_amdgcn_wmma_f32_16x16x32_f16(
          /*neg_a=*/false, a.v, /*neg_b=*/false, b.v,
          /*c_mod=*/(short)0, acc[nt], /*reuse_a=*/false, /*reuse_b=*/false);
    }
  }
  const int n = lane & 15;
  #pragma unroll
  for (int nt = 0; nt < 4; ++nt) {
    float bb = bias[nt * 16 + n];
    #pragma unroll
    for (int v = 0; v < 8; ++v) {
      long long r = rowBase + v + hi * 8;   // 16x16 f32 D layout: M = v + 8*hi, N = lane&15
      if (r < nRows) {
        float val = acc[nt][v] + bb;
        out[r * 64 + nt * 16 + n] = val > 0.f ? val : 0.f;
      }
    }
  }
}

// messages = relu(concat(x[row[e]], seg[col[e]] - has_rev*eh[rev[e]]) @ Wm + bm)
__global__ void __launch_bounds__(256) k_edge_gemm(
    const float* __restrict__ x, int inx,
    const float* __restrict__ seg, const float* __restrict__ eh,
    const int* __restrict__ row, const int* __restrict__ col,
    const int* __restrict__ rev, const int* __restrict__ hasrev,
    const _Float16* __restrict__ Wg, const float* __restrict__ bias,
    float* __restrict__ out, long long E, int KC) {
  const int kpad = KC * 32;
  __shared__ __align__(32) _Float16 sW[WSLOT_HALF];
  __shared__ __align__(32) _Float16 sA[8 * 16 * KPAD_MAX];

  {   // cooperative weight load (already swizzled in global)
    const unsigned int* g = (const unsigned int*)Wg;
    unsigned int* s = (unsigned int*)sW;
    for (int i = threadIdx.x; i < kpad * 32; i += 256) s[i] = g[i];
  }
  const int wave = threadIdx.x >> 5, lane = threadIdx.x & 31;
  const long long base = (long long)blockIdx.x * 128 + wave * 16;
  _Float16* sAw = sA + wave * 16 * KPAD_MAX;

  for (int e = 0; e < 16; ++e) {
    long long eg = base + e;
    if (eg < E) {
      int r = row[eg], c = col[eg], rv = rev[eg], hr = hasrev[eg];
      const float* xr = x   + (long long)r * inx;
      const float* sc = seg + (long long)c * 64;
      const float* er = eh  + (long long)rv * 64;
      for (int k = lane; k < kpad; k += 32) {
        float v;
        if (k < inx) v = xr[k];
        else if (k < inx + 64) { int hh = k - inx; v = sc[hh] - (hr ? er[hh] : 0.f); }
        else v = 0.f;
        sAw[e * KPAD_MAX + k] = (_Float16)v;
      }
    } else {
      for (int k = lane; k < kpad; k += 32) sAw[e * KPAD_MAX + k] = (_Float16)0.f;
    }
  }
  __syncthreads();
  wmma_rows16(sAw, sW, KC, bias, out, base, E, lane);
}

// x_out = relu(concat(x, node_msg) @ Wn + bn)
__global__ void __launch_bounds__(256) k_node_gemm(
    const float* __restrict__ x, int inx, const float* __restrict__ nodemsg,
    const _Float16* __restrict__ Wg, const float* __restrict__ bias,
    float* __restrict__ out, long long N, int KC) {
  const int kpad = KC * 32;
  __shared__ __align__(32) _Float16 sW[WSLOT_HALF];
  __shared__ __align__(32) _Float16 sA[8 * 16 * KPAD_MAX];
  {
    const unsigned int* g = (const unsigned int*)Wg;
    unsigned int* s = (unsigned int*)sW;
    for (int i = threadIdx.x; i < kpad * 32; i += 256) s[i] = g[i];
  }
  const int wave = threadIdx.x >> 5, lane = threadIdx.x & 31;
  const long long base = (long long)blockIdx.x * 128 + wave * 16;
  _Float16* sAw = sA + wave * 16 * KPAD_MAX;

  for (int e = 0; e < 16; ++e) {
    long long ng = base + e;
    if (ng < N) {
      const float* xr = x + ng * inx;
      const float* nm = nodemsg + ng * 64;
      for (int k = lane; k < kpad; k += 32) {
        float v;
        if (k < inx) v = xr[k];
        else if (k < inx + 64) v = nm[k - inx];
        else v = 0.f;
        sAw[e * KPAD_MAX + k] = (_Float16)v;
      }
    } else {
      for (int k = lane; k < kpad; k += 32) sAw[e * KPAD_MAX + k] = (_Float16)0.f;
    }
  }
  __syncthreads();
  wmma_rows16(sAw, sW, KC, bias, out, base, N, lane);
}

// ---------------- tiny MLP head (B=512, ~0.5 GFLOP: plain f32) ----------------

__global__ void k_xg(const float* __restrict__ pooled, const float* __restrict__ Wxd,
                     const float* __restrict__ bxd, float* __restrict__ xg, int B) {
  int b = blockIdx.x; int t = threadIdx.x;       // blockDim = 128
  if (b >= B) return;
  float s = bxd[t];
  const float* p = pooled + (long long)b * 64;
  #pragma unroll 8
  for (int k = 0; k < 64; ++k) s += p[k] * Wxd[k * 128 + t];
  xg[(long long)b * 128 + t] = s > 0.f ? s : 0.f;
}

__global__ void k_h1(const float* __restrict__ xg, const float* __restrict__ xt,
                     const float* __restrict__ W1, const float* __restrict__ b1,
                     float* __restrict__ h1, int B) {
  long long t = (long long)blockIdx.x * blockDim.x + threadIdx.x;
  if (t >= (long long)B * 1024) return;
  int b = (int)(t >> 10), c = (int)(t & 1023);
  float s = b1[c];
  const float* g = xg + (long long)b * 128;
  const float* tt = xt + (long long)b * 128;
  for (int j = 0; j < 128; ++j) s += g[j] * W1[j * 1024 + c];
  for (int j = 0; j < 128; ++j) s += tt[j] * W1[(128 + j) * 1024 + c];
  h1[t] = s > 0.f ? s : 0.f;
}

__global__ void k_h2(const float* __restrict__ h1, const float* __restrict__ W2,
                     const float* __restrict__ b2, float* __restrict__ h2, int B) {
  long long t = (long long)blockIdx.x * blockDim.x + threadIdx.x;
  if (t >= (long long)B * 256) return;
  int b = (int)(t >> 8), c = (int)(t & 255);
  float s = b2[c];
  const float* h = h1 + (long long)b * 1024;
  for (int j = 0; j < 1024; ++j) s += h[j] * W2[j * 256 + c];
  h2[t] = s > 0.f ? s : 0.f;
}

__global__ void k_out(const float* __restrict__ h2, const float* __restrict__ Wo,
                      const float* __restrict__ bo, float* __restrict__ out, int B) {
  int b = blockIdx.x * blockDim.x + threadIdx.x;
  if (b >= B) return;
  float s = bo[0];
  const float* h = h2 + (long long)b * 256;
  for (int j = 0; j < 256; ++j) s += h[j] * Wo[j];
  out[b] = s;
}

// ---------------- host orchestration ----------------

extern "C" void kernel_launch(void* const* d_in, const int* in_sizes, int n_in,
                              void* d_out, int out_size, void* d_ws, size_t ws_size,
                              hipStream_t stream) {
  (void)n_in; (void)out_size; (void)ws_size;
  const float* x    = (const float*)d_in[0];
  const float* ea   = (const float*)d_in[1];
  const float* xt   = (const float*)d_in[2];
  const int*   eidx = (const int*)d_in[3];
  const int*   batch= (const int*)d_in[4];
  const float* We = (const float*)d_in[6];  const float* be = (const float*)d_in[7];
  const float* Wm[3] = { (const float*)d_in[8],  (const float*)d_in[12], (const float*)d_in[16] };
  const float* bm[3] = { (const float*)d_in[9],  (const float*)d_in[13], (const float*)d_in[17] };
  const float* Wn[3] = { (const float*)d_in[10], (const float*)d_in[14], (const float*)d_in[18] };
  const float* bn[3] = { (const float*)d_in[11], (const float*)d_in[15], (const float*)d_in[19] };
  const float* Wxd = (const float*)d_in[20]; const float* bxd = (const float*)d_in[21];
  const float* W1  = (const float*)d_in[22]; const float* b1  = (const float*)d_in[23];
  const float* W2  = (const float*)d_in[24]; const float* b2  = (const float*)d_in[25];
  const float* Wo  = (const float*)d_in[26]; const float* bo  = (const float*)d_in[27];

  const long long N = in_sizes[0] / 78;
  const long long E = in_sizes[3] / 2;
  const int       B = in_sizes[2] / 128;
  const int* row = eidx;
  const int* col = eidx + E;

  long long hsll = 1; while (hsll < 2 * E) hsll <<= 1;
  const int hs = (int)hsll;
  const unsigned int mask = (unsigned int)(hs - 1);

  char* ws = (char*)d_ws;
  size_t off = 0;
  auto take = [&](size_t bytes) -> void* {
    void* p = ws + off;
    off = (off + bytes + 255) & ~(size_t)255;
    return p;
  };
  int* rev   = (int*)take((size_t)E * 4);
  int* hrev  = (int*)take((size_t)E * 4);
  unsigned long long* hkeys = (unsigned long long*)take((size_t)hs * 8);
  int* hvals = (int*)take((size_t)hs * 4);
  float* ehA = (float*)take((size_t)E * 64 * 4);
  float* ehB = (float*)take((size_t)E * 64 * 4);
  float* seg = (float*)take((size_t)N * 64 * 4);
  float* nodemsg = (float*)take((size_t)N * 64 * 4);
  float* xb1 = (float*)take((size_t)N * 64 * 4);
  float* xb2 = (float*)take((size_t)N * 64 * 4);
  _Float16* wslots = (_Float16*)take((size_t)6 * WSLOT_HALF * 2);
  float* pooled = (float*)take((size_t)B * 64 * 4);
  float* xg = (float*)take((size_t)B * 128 * 4);
  float* h1 = (float*)take((size_t)B * 1024 * 4);
  float* h2 = (float*)take((size_t)B * 256 * 4);

  // --- reverse edge map ---
  k_hash_clear<<<(hs + 255) / 256, 256, 0, stream>>>(hkeys, hs);
  k_hash_insert<<<(int)((E + 255) / 256), 256, 0, stream>>>(row, col, E, N, hkeys, hvals, mask);
  k_hash_lookup<<<(int)((E + 255) / 256), 256, 0, stream>>>(row, col, E, N, hkeys, hvals, mask, rev, hrev);

  // --- weight prep (f16, WMMA-swizzled) ---
  const int Kin[3] = { 78 + 64, 64 + 64, 64 + 64 };
  for (int l = 0; l < 3; ++l) {
    int KC = (Kin[l] + 31) / 32;
    int tot = KC * 2048;
    k_prep_w<<<(tot + 255) / 256, 256, 0, stream>>>(Wm[l], Kin[l], KC, wslots + (size_t)(2 * l)     * WSLOT_HALF);
    k_prep_w<<<(tot + 255) / 256, 256, 0, stream>>>(Wn[l], Kin[l], KC, wslots + (size_t)(2 * l + 1) * WSLOT_HALF);
  }

  // --- edge init ---
  k_edge_init<<<(int)((E * 64 + 255) / 256), 256, 0, stream>>>(ea, We, be, ehA, E);

  // --- 3 DMPNN layers ---
  const float* xin = x; int inx = 78;
  float* ehCur = ehA; float* ehNxt = ehB;
  for (int l = 0; l < 3; ++l) {
    int KC = (inx + 64 + 31) / 32;
    k_zero_f32<<<(int)((N * 64 + 255) / 256), 256, 0, stream>>>(seg, N * 64);
    k_segsum64<<<(int)((E * 64 + 255) / 256), 256, 0, stream>>>(ehCur, row, seg, E);
    k_edge_gemm<<<(int)((E + 127) / 128), 256, 0, stream>>>(
        xin, inx, seg, ehCur, row, col, rev, hrev,
        wslots + (size_t)(2 * l) * WSLOT_HALF, bm[l], ehNxt, E, KC);
    k_zero_f32<<<(int)((N * 64 + 255) / 256), 256, 0, stream>>>(nodemsg, N * 64);
    k_segsum64<<<(int)((E * 64 + 255) / 256), 256, 0, stream>>>(ehNxt, col, nodemsg, E);
    float* xout = (l == 1) ? xb2 : xb1;
    k_node_gemm<<<(int)((N + 127) / 128), 256, 0, stream>>>(
        xin, inx, nodemsg, wslots + (size_t)(2 * l + 1) * WSLOT_HALF, bn[l], xout, N, KC);
    { float* t = ehCur; ehCur = ehNxt; ehNxt = t; }
    xin = xout; inx = 64;
  }

  // --- pooling + head ---
  k_zero_f32<<<(B * 64 + 255) / 256, 256, 0, stream>>>(pooled, (long long)B * 64);
  k_segsum64<<<(int)((N * 64 + 255) / 256), 256, 0, stream>>>(xin, batch, pooled, N);
  k_xg<<<B, 128, 0, stream>>>(pooled, Wxd, bxd, xg, B);
  k_h1<<<(B * 1024 + 255) / 256, 256, 0, stream>>>(xg, xt, W1, b1, h1, B);
  k_h2<<<(B * 256 + 255) / 256, 256, 0, stream>>>(h1, W2, b2, h2, B);
  k_out<<<(B + 255) / 256, 256, 0, stream>>>(h2, Wo, bo, (float*)d_out, B);
}
